// VIT_14250701488151
// MI455X (gfx1250) — matrix operand order repaired and checked by reading, not compiled, not run on hardware
//
#include <hip/hip_runtime.h>
#include <math.h>

// ---------------- CDNA5 WMMA types ----------------
typedef __attribute__((ext_vector_type(16))) __bf16 v16bf;
typedef __attribute__((ext_vector_type(8)))  __bf16 v8bf;
typedef __attribute__((ext_vector_type(8)))  float  v8f;

// ---------------- problem constants ----------------
static constexpr int kB     = 16;
static constexpr int kNSeq  = 481;   // 12*40 patches + cls
static constexpr int kD     = 384;
static constexpr int kHeads = 6;
static constexpr int kDh    = 64;
static constexpr int kFF    = 1536;
static constexpr int kL     = 12;
static constexpr int kNP    = 480;
static constexpr int kRows  = kB * kNSeq;   // 7696
static constexpr int kPRows = kB * kNP;     // 7680
static constexpr int kLdAtt = 512;          // padded score / K^T ld
static constexpr int kSeqP  = 496;          // padded sequence rows per (b,h) score block

// =====================================================================
// Branchless batched WMMA GEMM:  C = act(alpha * A*B + bias) (+resid)
//   A: f32 [M x K] row-major (lda); reads are in-bounds by allocation padding
//   B: [K x N] row-major (ldb); BT: 1 = bf16 packed, 0 = f32
//   K must be a multiple of 32. Store-side guards only.
// =====================================================================
template <int BT>
__global__ __launch_bounds__(32)
void vit_gemm_wmma(const float* __restrict__ A,
                   const void* __restrict__ Bv,
                   const float* __restrict__ bias,
                   const float* __restrict__ resid,
                   float* __restrict__ C,
                   int M, int N, int K,
                   int lda, int ldb, int ldc,
                   long long sA1, long long sA2,
                   long long sB1, long long sB2,
                   long long sC1, long long sC2,
                   int zdiv, float alpha, int act)
{
    const int lane = threadIdx.x;
    const int z    = blockIdx.z;
    const int zq   = z / zdiv;
    const int zr   = z % zdiv;
    const long long offA = (long long)zq * sA1 + (long long)zr * sA2;
    const long long offB = (long long)zq * sB1 + (long long)zr * sB2;
    const long long offC = (long long)zq * sC1 + (long long)zr * sC2;

    const int tm = blockIdx.x * 16;
    const int tn = blockIdx.y * 64;

    v8f accs[4];
#pragma unroll
    for (int nt = 0; nt < 4; ++nt)
#pragma unroll
        for (int r = 0; r < 8; ++r) accs[nt][r] = 0.0f;

    const int half = lane >> 4;            // 0: lanes 0-15, 1: lanes 16-31
    const int mrow = tm + (lane & 15);
    const float*  Arow = A + offA + (long long)mrow * lda;
    const float*  Bf = (const float*)Bv;
    const __bf16* Bh = (const __bf16*)Bv;

    for (int k0 = 0; k0 < K; k0 += 32) {
        // ---- A fragment (ISA 7.12.2): two contiguous 8-float runs per lane
        const float4* pa = (const float4*)(Arow + k0 + (half << 3));
        float4 a0 = pa[0], a1 = pa[1], a2 = pa[4], a3 = pa[5];
        v16bf af;
        af[0]  = (__bf16)a0.x; af[1]  = (__bf16)a0.y; af[2]  = (__bf16)a0.z; af[3]  = (__bf16)a0.w;
        af[4]  = (__bf16)a1.x; af[5]  = (__bf16)a1.y; af[6]  = (__bf16)a1.z; af[7]  = (__bf16)a1.w;
        af[8]  = (__bf16)a2.x; af[9]  = (__bf16)a2.y; af[10] = (__bf16)a2.z; af[11] = (__bf16)a2.w;
        af[12] = (__bf16)a3.x; af[13] = (__bf16)a3.y; af[14] = (__bf16)a3.z; af[15] = (__bf16)a3.w;

        // ---- B fragments: lane = N column, K in table order (rows strided by ldb) ----
        const long long brow = offB + (long long)(k0 + (half << 3)) * ldb + (lane & 15);
#pragma unroll
        for (int nt = 0; nt < 4; ++nt) {
            const int col16 = tn + nt * 16;
            v16bf bf;
            if (BT) {
                const __bf16* bp = Bh + brow + col16;
#pragma unroll
                for (int j = 0; j < 8; ++j) { bf[j] = bp[(long long)j * ldb]; bf[8 + j] = bp[(long long)(16 + j) * ldb]; }
            } else {
                const float* bp = Bf + brow + col16;
#pragma unroll
                for (int j = 0; j < 8; ++j) { bf[j] = (__bf16)bp[(long long)j * ldb]; bf[8 + j] = (__bf16)bp[(long long)(16 + j) * ldb]; }
            }
            accs[nt] = __builtin_amdgcn_wmma_f32_16x16x32_bf16(
                false, af, false, bf, (short)0, accs[nt], false, false);
        }
    }

    // ---- epilogue: alpha, bias, GELU(exact), residual; store guards only ----
#pragma unroll
    for (int nt = 0; nt < 4; ++nt) {
        int col = tn + nt * 16 + (lane & 15);
        if (col >= N) continue;
        float bv = bias ? bias[col] : 0.0f;
#pragma unroll
        for (int r = 0; r < 8; ++r) {
            int row = tm + r + half * 8;
            if (row < M) {
                float v = accs[nt][r] * alpha + bv;
                if (act == 1) v = 0.5f * v * (1.0f + erff(v * 0.70710678118654752f));
                long long ci = offC + (long long)row * ldc + col;
                if (resid) v += resid[ci];
                C[ci] = v;
            }
        }
    }
}

// =====================================================================
// Weight pack: W [N x K] f32 row-major -> WT [K x N] bf16 row-major
// =====================================================================
__global__ void vit_pack_w(const float* __restrict__ W, __bf16* __restrict__ WT,
                           int N, int K)
{
    long long idx = (long long)blockIdx.x * blockDim.x + threadIdx.x;
    if (idx >= (long long)N * K) return;
    int n = (int)(idx % N);
    int k = (int)(idx / N);
    WT[(long long)k * N + n] = (__bf16)W[(long long)n * K + k];
}

// attention K pack: KT[bh][k(64) x n(481)] bf16, ld = kLdAtt
__global__ void vit_pack_kt(const float* __restrict__ QKV, __bf16* __restrict__ KT)
{
    long long idx = (long long)blockIdx.x * blockDim.x + threadIdx.x;
    if (idx >= (long long)kB * kHeads * kDh * kNSeq) return;
    int n  = (int)(idx % kNSeq);
    int k  = (int)((idx / kNSeq) % kDh);
    int bh = (int)(idx / ((long long)kNSeq * kDh));
    int b = bh / kHeads, h = bh % kHeads;
    KT[((long long)bh * kDh + k) * kLdAtt + n] =
        (__bf16)QKV[((long long)b * kNSeq + n) * (3 * kD) + kD + h * kDh + k];
}

// final-stage K pack: K2T[b][k(384) x n(480)] bf16, ld = kLdAtt
__global__ void vit_pack_k2t(const float* __restrict__ QKV2, __bf16* __restrict__ KT)
{
    long long idx = (long long)blockIdx.x * blockDim.x + threadIdx.x;
    if (idx >= (long long)kB * kD * kNP) return;
    int n = (int)(idx % kNP);
    int k = (int)((idx / kNP) % kD);
    int b = (int)(idx / ((long long)kNP * kD));
    KT[((long long)b * kD + k) * kLdAtt + n] =
        (__bf16)QKV2[((long long)b * kNP + n) * (3 * kD) + kD + k];
}

// =====================================================================
// LayerNorm over D=384, one block per row
// =====================================================================
__global__ __launch_bounds__(128)
void vit_ln(const float* __restrict__ X, const float* __restrict__ w,
            const float* __restrict__ b, float* __restrict__ Y)
{
    __shared__ float red[128];
    const int t = threadIdx.x;
    const long long row = blockIdx.x;
    const float* x = X + row * kD;
    float* y = Y + row * kD;

    float v0 = x[t], v1 = x[t + 128], v2 = x[t + 256];
    red[t] = v0 + v1 + v2;
    __syncthreads();
    for (int o = 64; o > 0; o >>= 1) { if (t < o) red[t] += red[t + o]; __syncthreads(); }
    float mean = red[0] * (1.0f / kD);
    __syncthreads();

    float d0 = v0 - mean, d1 = v1 - mean, d2 = v2 - mean;
    red[t] = d0 * d0 + d1 * d1 + d2 * d2;
    __syncthreads();
    for (int o = 64; o > 0; o >>= 1) { if (t < o) red[t] += red[t + o]; __syncthreads(); }
    float rstd = rsqrtf(red[0] * (1.0f / kD) + 1e-6f);

    y[t]       = d0 * rstd * w[t]       + b[t];
    y[t + 128] = d1 * rstd * w[t + 128] + b[t + 128];
    y[t + 256] = d2 * rstd * w[t + 256] + b[t + 256];
}

// =====================================================================
// Row softmax (in place) + zero-fill of pad columns [ncols, ldn)
// rows indexed as bh-block (rowsPer rows, bstride apart, ld ldn)
// =====================================================================
__global__ __launch_bounds__(128)
void vit_softmax(float* __restrict__ S, int ncols, int ldn,
                 int rowsPer, long long bstride)
{
    __shared__ float red[128];
    const int t  = threadIdx.x;
    const int g  = blockIdx.x;
    const int bh = g / rowsPer;
    const int m  = g % rowsPer;
    float* row = S + (long long)bh * bstride + (long long)m * ldn;

    float mx = -3.4e38f;
    for (int c = t; c < ncols; c += 128) mx = fmaxf(mx, row[c]);
    red[t] = mx; __syncthreads();
    for (int o = 64; o > 0; o >>= 1) { if (t < o) red[t] = fmaxf(red[t], red[t + o]); __syncthreads(); }
    mx = red[0]; __syncthreads();

    float s = 0.0f;
    for (int c = t; c < ncols; c += 128) { float e = __expf(row[c] - mx); row[c] = e; s += e; }
    red[t] = s; __syncthreads();
    for (int o = 64; o > 0; o >>= 1) { if (t < o) red[t] += red[t + o]; __syncthreads(); }
    float inv = 1.0f / red[0];
    for (int c = t; c < ncols; c += 128) row[c] *= inv;
    for (int c = ncols + t; c < ldn; c += 128) row[c] = 0.0f;   // pad for K=512 P*V
}

// =====================================================================
// Patch embedding (16x16 stride-16 conv) + pos embed; cls row separate
// =====================================================================
__global__ void vit_patch_embed(const float* __restrict__ in, const float* __restrict__ cw,
                                const float* __restrict__ cb, const float* __restrict__ pos,
                                float* __restrict__ X)
{
    long long idx = (long long)blockIdx.x * blockDim.x + threadIdx.x;
    if (idx >= (long long)kPRows * kD) return;
    int d = (int)(idx % kD);
    long long mp = idx / kD;             // b*480 + p
    int b = (int)(mp / kNP), p = (int)(mp % kNP);
    int py = p / 40, px = p % 40;
    float acc = cb[d];
    for (int c = 0; c < 7; ++c)
        for (int iy = 0; iy < 16; ++iy) {
            const float4* i4 = (const float4*)(in + (((long long)(b * 7 + c) * 192 + py * 16 + iy) * 640 + px * 16));
            const float4* w4 = (const float4*)(cw + ((long long)(d * 7 + c) * 16 + iy) * 16);
#pragma unroll
            for (int q = 0; q < 4; ++q) {
                float4 a = i4[q], w = w4[q];
                acc = fmaf(a.x, w.x, acc);
                acc = fmaf(a.y, w.y, acc);
                acc = fmaf(a.z, w.z, acc);
                acc = fmaf(a.w, w.w, acc);
            }
        }
    X[((long long)b * kNSeq + 1 + p) * kD + d] = acc + pos[(long long)(1 + p) * kD + d];
}

__global__ void vit_cls(const float* __restrict__ cls, const float* __restrict__ pos,
                        float* __restrict__ X)
{
    int i = blockIdx.x * blockDim.x + threadIdx.x;
    if (i >= kB * kD) return;
    int b = i / kD, d = i % kD;
    X[(long long)b * kNSeq * kD + d] = cls[d] + pos[d];
}

// =====================================================================
// Final stage: per (b,q,r) group-of-12 softmax, lower-median mask,
// scatter to col = r + 40*c  (bijection on 0..479)
// =====================================================================
__global__ void vit_part(const float* __restrict__ ATT2, float* __restrict__ PART,
                         float* __restrict__ MASK)
{
    int idx = blockIdx.x * blockDim.x + threadIdx.x;   // kPRows * 40
    if (idx >= kPRows * 40) return;
    int r  = idx % 40;
    int aq = idx / 40;                                  // b*480 + q
    const float* row = ATT2 + (long long)aq * kNP;

    float v[12];
    float mx = -3.4e38f;
#pragma unroll
    for (int c = 0; c < 12; ++c) { v[c] = row[r + 40 * c]; mx = fmaxf(mx, v[c]); }
    float s = 0.0f;
#pragma unroll
    for (int c = 0; c < 12; ++c) { v[c] = __expf(v[c] - mx); s += v[c]; }
    float inv = 1.0f / s;
#pragma unroll
    for (int c = 0; c < 12; ++c) v[c] *= inv;

    float sv[12];
#pragma unroll
    for (int c = 0; c < 12; ++c) sv[c] = v[c];
    for (int i = 1; i < 12; ++i) {          // insertion sort ascending
        float key = sv[i]; int j = i - 1;
        while (j >= 0 && sv[j] > key) { sv[j + 1] = sv[j]; --j; }
        sv[j + 1] = key;
    }
    float med = sv[5];                       // index (12-1)//2

    float* prow = PART + (long long)aq * kNP;
    float* mrow = MASK + (long long)aq * kNP;
#pragma unroll
    for (int c = 0; c < 12; ++c) {
        float p = v[c];
        prow[r + 40 * c] = p;
        mrow[r + 40 * c] = (p > med ? 0.0f : p) * (1.0f / 24.0f);  // / (D//8/2)
    }
}

__global__ void vit_colmean(const float* __restrict__ Min, float* __restrict__ outp)
{
    int idx = blockIdx.x * blockDim.x + threadIdx.x;   // kB * 480
    if (idx >= kB * kNP) return;
    int b = idx / kNP, col = idx % kNP;
    const float* base = Min + (long long)b * kNP * kNP + col;
    float s = 0.0f;
    for (int q = 0; q < kNP; ++q) s += base[(long long)q * kNP];
    outp[idx] = s * (1.0f / kNP);
}

__global__ void vit_rowsum(const float* __restrict__ PART, float* __restrict__ RS)
{
    int idx = blockIdx.x * blockDim.x + threadIdx.x;   // kPRows
    if (idx >= kPRows) return;
    const float* row = PART + (long long)idx * kNP;
    float s = 0.0f;
    for (int c = 0; c < kNP; ++c) s += row[c];
    RS[idx] = s;
}

__global__ void vit_vsum(const float* __restrict__ QKV2, float* __restrict__ VS)
{
    int idx = blockIdx.x * blockDim.x + threadIdx.x;   // kB * kD
    if (idx >= kB * kD) return;
    int b = idx / kD, c = idx % kD;
    const float* base = QKV2 + (long long)b * kNP * (3 * kD) + 2 * kD + c;
    float s = 0.0f;
    for (int n = 0; n < kNP; ++n) s += base[(long long)n * (3 * kD)];
    VS[idx] = s;
}

__global__ void vit_outmat(const float* __restrict__ RS, const float* __restrict__ VS,
                           float* __restrict__ OUTM)
{
    long long idx = (long long)blockIdx.x * blockDim.x + threadIdx.x;  // kPRows*kD
    if (idx >= (long long)kPRows * kD) return;
    int c = (int)(idx % kD);
    long long row = idx / kD;
    int b = (int)(row / kNP);
    OUTM[idx] = RS[row] * VS[b * kD + c];
}

__global__ void vit_meanrows(const float* __restrict__ OUT3, float* __restrict__ out0)
{
    int idx = blockIdx.x * blockDim.x + threadIdx.x;   // kB * kD
    if (idx >= kB * kD) return;
    int b = idx / kD, c = idx % kD;
    const float* base = OUT3 + (long long)b * kNP * kD + c;
    float s = 0.0f;
    for (int a = 0; a < kNP; ++a) s += base[(long long)a * kD];
    out0[idx] = s * (1.0f / kNP);
}

// =====================================================================
// host side
// =====================================================================
static void launch_gemm(const float* A, const void* Bm, const float* bias,
                        const float* resid, float* C,
                        int M, int N, int K, int lda, int ldb, int ldc,
                        long long sA1, long long sA2, long long sB1, long long sB2,
                        long long sC1, long long sC2, int zdiv, int Z,
                        float alpha, int btype, int act, hipStream_t stream)
{
    dim3 grid((unsigned)((M + 15) / 16), (unsigned)((N + 63) / 64), (unsigned)Z);
    if (btype)
        vit_gemm_wmma<1><<<grid, dim3(32, 1, 1), 0, stream>>>(
            A, Bm, bias, resid, C, M, N, K, lda, ldb, ldc,
            sA1, sA2, sB1, sB2, sC1, sC2, zdiv, alpha, act);
    else
        vit_gemm_wmma<0><<<grid, dim3(32, 1, 1), 0, stream>>>(
            A, Bm, bias, resid, C, M, N, K, lda, ldb, ldc,
            sA1, sA2, sB1, sB2, sC1, sC2, zdiv, alpha, act);
}

static void launch_pack_w(const float* W, __bf16* WT, int N, int K, hipStream_t stream)
{
    long long tot = (long long)N * K;
    vit_pack_w<<<(unsigned)((tot + 255) / 256), 256, 0, stream>>>(W, WT, N, K);
}

extern "C" void kernel_launch(void* const* d_in, const int* in_sizes, int n_in,
                              void* d_out, int out_size, void* d_ws, size_t ws_size,
                              hipStream_t stream)
{
    (void)in_sizes; (void)n_in; (void)out_size; (void)ws_size;

    const float* input    = (const float*)d_in[0];
    const float* conv_w   = (const float*)d_in[1];
    const float* conv_b   = (const float*)d_in[2];
    const float* cls_tok  = (const float*)d_in[3];
    const float* pos_emb  = (const float*)d_in[4];
    const float* ln1_w    = (const float*)d_in[5];
    const float* ln1_b    = (const float*)d_in[6];
    const float* qkv_w    = (const float*)d_in[7];
    const float* qkv_b    = (const float*)d_in[8];
    const float* proj_w   = (const float*)d_in[9];
    const float* proj_b   = (const float*)d_in[10];
    const float* ln2_w    = (const float*)d_in[11];
    const float* ln2_b    = (const float*)d_in[12];
    const float* fc1_w    = (const float*)d_in[13];
    const float* fc1_b    = (const float*)d_in[14];
    const float* fc2_w    = (const float*)d_in[15];
    const float* fc2_b    = (const float*)d_in[16];
    const float* norm_w   = (const float*)d_in[17];
    const float* norm_b   = (const float*)d_in[18];
    const float* to_qkv_w = (const float*)d_in[19];
    const float* pose_w   = (const float*)d_in[20];
    const float* pose_b   = (const float*)d_in[21];

    float* ws = (float*)d_ws;
    // ---- workspace layout (float units) ----
    float* X    = ws;                              // 7696*384
    float* H    = ws + 2955264LL;                  // 7696*384
    float* QKV  = ws + 5910528LL;                  // 7728*1152 (32 pad rows for K=512 P*V)
    float* O    = ws + 14813184LL;                 // 7696*384
    float* SHR  = ws + 17768448LL;                 // 96*496*512 scores / FF hidden / part / mask
    float* ATT  = SHR;
    float* FFH  = SHR;
    float* PART = SHR + 4000000LL;
    float* MASKB= SHR + 8000000LL;
    __bf16* WB16 = (__bf16*)(ws + 42147840LL);     // 21,823,488 bf16
    __bf16* KT   = (__bf16*)(ws + 53059584LL);     // 3,145,728 bf16 (layer K^T / final K^T)
    float* RS   = ws + 54632448LL;                 // 7680
    float* VS   = RS + kPRows;                     // 6144
    float* out  = (float*)d_out;

    // ---- bf16 [K x N] packed weights ----
    __bf16* WTqkv  = WB16;                          // 12 * 384*1152
    __bf16* WTproj = WTqkv  + 12LL * kD * (3 * kD); // 12 * 384*384
    __bf16* WTfc1  = WTproj + 12LL * kD * kD;       // 12 * 384*1536
    __bf16* WTfc2  = WTfc1  + 12LL * kD * kFF;      // 12 * 1536*384
    __bf16* WTtoq  = WTfc2  + 12LL * kFF * kD;      // 384*1152
    __bf16* WTpose = WTtoq  + (long long)kD * (3 * kD); // 384*384

    for (int i = 0; i < kL; ++i) {
        launch_pack_w(qkv_w  + (long long)i * 3 * kD * kD, WTqkv  + (long long)i * kD * 3 * kD, 3 * kD, kD, stream);
        launch_pack_w(proj_w + (long long)i * kD * kD,     WTproj + (long long)i * kD * kD,     kD,     kD, stream);
        launch_pack_w(fc1_w  + (long long)i * kFF * kD,    WTfc1  + (long long)i * kD * kFF,    kFF,    kD, stream);
        launch_pack_w(fc2_w  + (long long)i * kD * kFF,    WTfc2  + (long long)i * kFF * kD,    kD,     kFF, stream);
    }
    launch_pack_w(to_qkv_w, WTtoq,  3 * kD, kD, stream);
    launch_pack_w(pose_w,   WTpose, kD,     kD, stream);

    // ---- patch embedding + cls + pos ----
    {
        long long tot = (long long)kPRows * kD;
        vit_patch_embed<<<(unsigned)((tot + 255) / 256), 256, 0, stream>>>(
            input, conv_w, conv_b, pos_emb, X);
        vit_cls<<<(kB * kD + 255) / 256, 256, 0, stream>>>(cls_tok, pos_emb, X);
    }

    const long long sBH_qkv = (long long)kNSeq * 3 * kD;     // (b) stride in QKV
    const long long sATT_h  = (long long)kSeqP * kLdAtt;     // per-head score block (496x512)
    const long long sATT_b  = (long long)kHeads * sATT_h;
    const long long sKT_h   = (long long)kDh * kLdAtt;       // per-head K^T block
    const long long sKT_b   = (long long)kHeads * sKT_h;

    for (int i = 0; i < kL; ++i) {
        // LN1
        vit_ln<<<kRows, 128, 0, stream>>>(X, ln1_w + i * kD, ln1_b + i * kD, H);
        // QKV = H @ qkv_w^T + qkv_b
        launch_gemm(H, WTqkv + (long long)i * kD * 3 * kD, qkv_b + (long long)i * 3 * kD,
                    nullptr, QKV, kRows, 3 * kD, kD, kD, 3 * kD, 3 * kD,
                    0, 0, 0, 0, 0, 0, 1, 1, 1.0f, 1, 0, stream);
        // pack K^T (per b,h) to bf16
        {
            long long tot = (long long)kB * kHeads * kDh * kNSeq;
            vit_pack_kt<<<(unsigned)((tot + 255) / 256), 256, 0, stream>>>(QKV, KT);
        }
        // S = (q * dh^-0.5) @ k^T   per (b,h)
        launch_gemm(QKV, KT, nullptr, nullptr, ATT,
                    kNSeq, kNSeq, kDh, 3 * kD, kLdAtt, kLdAtt,
                    sBH_qkv, kDh, sKT_b, sKT_h, sATT_b, sATT_h,
                    kHeads, kB * kHeads, 0.125f, 1, 0, stream);
        // softmax rows (+ zero pad cols 481..511)
        vit_softmax<<<kB * kHeads * kNSeq, 128, 0, stream>>>(ATT, kNSeq, kLdAtt, kNSeq, sATT_h);
        // O = P @ V   per (b,h)   (V f32 [K x N] row-major; K padded to 512)
        launch_gemm(ATT, QKV + 2 * kD, nullptr, nullptr, O,
                    kNSeq, kDh, kLdAtt, kLdAtt, 3 * kD, kD,
                    sATT_b, sATT_h, sBH_qkv, kDh, (long long)kNSeq * kD, kDh,
                    kHeads, kB * kHeads, 1.0f, 0, 0, stream);
        // X += O @ proj_w^T + proj_b
        launch_gemm(O, WTproj + (long long)i * kD * kD, proj_b + (long long)i * kD,
                    X, X, kRows, kD, kD, kD, kD, kD,
                    0, 0, 0, 0, 0, 0, 1, 1, 1.0f, 1, 0, stream);
        // LN2
        vit_ln<<<kRows, 128, 0, stream>>>(X, ln2_w + i * kD, ln2_b + i * kD, H);
        // FFH = gelu(H @ fc1_w^T + fc1_b)
        launch_gemm(H, WTfc1 + (long long)i * kD * kFF, fc1_b + (long long)i * kFF,
                    nullptr, FFH, kRows, kFF, kD, kD, kFF, kFF,
                    0, 0, 0, 0, 0, 0, 1, 1, 1.0f, 1, 1, stream);
        // X += FFH @ fc2_w^T + fc2_b
        launch_gemm(FFH, WTfc2 + (long long)i * kFF * kD, fc2_b + (long long)i * kD,
                    X, X, kRows, kD, kFF, kFF, kD, kD,
                    0, 0, 0, 0, 0, 0, 1, 1, 1.0f, 1, 0, stream);
    }

    // ---- final norm ----
    vit_ln<<<kRows, 128, 0, stream>>>(X, norm_w, norm_b, H);

    // qkv2 = t @ to_qkv_w^T  (t drops cls row; per-batch offset)
    launch_gemm(H + kD, WTtoq, nullptr, nullptr, QKV,
                kNP, 3 * kD, kD, kD, 3 * kD, 3 * kD,
                (long long)kNSeq * kD, 0, 0, 0, (long long)kNP * 3 * kD, 0,
                1, kB, 1.0f, 1, 0, stream);
    // pack final K^T  [b][384 x 480] bf16, ld 512
    {
        long long tot = (long long)kB * kD * kNP;
        vit_pack_k2t<<<(unsigned)((tot + 255) / 256), 256, 0, stream>>>(QKV, KT);
    }
    // attn2 = (q*0.05103) @ k^T / 100   per batch
    launch_gemm(QKV, KT, nullptr, nullptr, ATT,
                kNP, kNP, kD, 3 * kD, kLdAtt, kNP,
                (long long)kNP * 3 * kD, 0, (long long)kD * kLdAtt, 0,
                (long long)kNP * kNP, 0,
                1, kB, 0.0005103f, 1, 0, stream);

    // group softmax + median mask + scatter
    vit_part<<<(kPRows * 40 + 255) / 256, 256, 0, stream>>>(ATT, PART, MASKB);
    // attn_map / masked_map (column means over q)
    vit_colmean<<<(kB * kNP + 255) / 256, 256, 0, stream>>>(PART, out + kB * kD);
    vit_colmean<<<(kB * kNP + 255) / 256, 256, 0, stream>>>(MASKB, out + kB * kD + kB * kNP);
    // row/col sums for the factorized einsum
    vit_rowsum<<<(kPRows + 255) / 256, 256, 0, stream>>>(PART, RS);
    vit_vsum<<<(kB * kD + 255) / 256, 256, 0, stream>>>(QKV, VS);
    {
        long long tot = (long long)kPRows * kD;
        vit_outmat<<<(unsigned)((tot + 255) / 256), 256, 0, stream>>>(RS, VS, O);
    }
    // out2 = out @ pose_w^T + pose_b   (into FFH region; PART/MASK consumed)
    launch_gemm(O, WTpose, pose_b, nullptr, FFH,
                kPRows, kD, kD, kD, kD, kD,
                0, 0, 0, 0, 0, 0, 1, 1, 1.0f, 1, 0, stream);
    // final LN then mean over sequence
    vit_ln<<<kPRows, 128, 0, stream>>>(FFH, norm_w, norm_b, H);
    vit_meanrows<<<(kB * kD + 255) / 256, 256, 0, stream>>>(H, out);
}
